// MyNet_51333449121964
// MI455X (gfx1250) — matrix operand order repaired and checked by reading, not compiled, not run on hardware
//
#include <hip/hip_runtime.h>
#include <hip/hip_bf16.h>
#include <stdint.h>

// ---------- types ----------
typedef __bf16 bf16_t;
typedef __attribute__((ext_vector_type(16))) __bf16 v16bf;
typedef __attribute__((ext_vector_type(8)))  float  v8f;
typedef unsigned int u32x4 __attribute__((ext_vector_type(4)));
typedef unsigned int u32x2 __attribute__((ext_vector_type(2)));

union Frag { u32x4 q[2]; v16bf v; };

// ---------- graph normalization ----------
__global__ void init_deg_kernel(float* deg, int n) {
    int i = blockIdx.x * blockDim.x + threadIdx.x;
    if (i < n) deg[i] = 1.0f;  // self loop contributes 1 to every node
}

__global__ void deg_kernel(const long long* __restrict__ dst, float* deg, int e) {
    int i = blockIdx.x * blockDim.x + threadIdx.x;
    if (i < e) atomicAdd(&deg[(int)dst[i]], 1.0f);
}

__global__ void dinv_kernel(const float* __restrict__ deg, float* dinv, int n) {
    int i = blockIdx.x * blockDim.x + threadIdx.x;
    if (i < n) {
        float d = deg[i];
        dinv[i] = (d > 0.0f) ? rsqrtf(d) : 0.0f;
    }
}

// ---------- precision conversion ----------
__global__ void convert_x_kernel(const float* __restrict__ x, bf16_t* __restrict__ xb, int total) {
    int i = blockIdx.x * blockDim.x + threadIdx.x;
    if (i < total) xb[i] = (bf16_t)x[i];
}

// W is [K][Fout] row-major; produce Wt [FoutPad][K] bf16, zero-padding cols >= Fout.
__global__ void convert_w_kernel(const float* __restrict__ W, bf16_t* __restrict__ Wt,
                                 int K, int Fout, int FoutPad) {
    int i = blockIdx.x * blockDim.x + threadIdx.x;
    if (i >= FoutPad * K) return;
    int o = i / K;
    int k = i - o * K;
    Wt[i] = (o < Fout) ? (bf16_t)W[(size_t)k * Fout + o] : (bf16_t)0.0f;
}

// ---------- WMMA GEMM: H[N x FoutPad](bf16) = A[N x K](bf16) * Wt^T ----------
// Each wave computes a 16 x (CT*16) strip: one A fragment reused across CT B
// tiles per K-step. All CT B fragments are loaded up front each K-step so the
// loads clause together and the compiler can issue WMMAs under partial
// s_wait_loadcnt instead of serializing each WMMA behind its own load.
//
// A fragment (16-bit A 16x32 layout): lane l -> row = l&15, kbase = (l>>4)*16;
//   elems j = K kbase + j (contiguous, matching B).
// B fragment (16-bit B 32x16 layout): lane l -> col = l&15;
//   elems j = K (l>>4)*16 + j  (contiguous in Wt row).
// C/D: lane l, vgpr r -> row = 8*(l>>4)+r, col = l&15.
template <int CT>
__global__ __launch_bounds__(256)
void gemm_wmma_kernel(const bf16_t* __restrict__ A, const bf16_t* __restrict__ Wt,
                      bf16_t* __restrict__ H, int nRowTiles, int nColGroups,
                      int K, int ldH) {
    int gwave = (blockIdx.x * blockDim.x + threadIdx.x) >> 5;
    int lane  = threadIdx.x & 31;
    int tM = gwave / nColGroups;
    int tg = gwave - tM * nColGroups;
    if (tM >= nRowTiles) return;  // wave-uniform: EXEC stays all-1s for WMMA

    int r  = lane & 15;
    int hi = lane >> 4;

    const bf16_t* aRow  = A  + ((size_t)(tM * 16 + r)) * K + hi * 16;         // 32B aligned
    const bf16_t* bRow0 = Wt + ((size_t)(tg * CT * 16 + r)) * K + hi * 16;    // 32B aligned

    v8f acc[CT];
    #pragma unroll
    for (int c = 0; c < CT; ++c) acc[c] = (v8f){};

    for (int k0 = 0; k0 < K; k0 += 32) {
        // Issue all loads for this K-step first (clause-friendly).
        Frag a;
        const u32x4* pa = (const u32x4*)(aRow + k0);
        a.q[0] = pa[0];   // K = kbase + 0..7
        a.q[1] = pa[1];   // K = kbase + 8..15
        Frag b[CT];
        #pragma unroll
        for (int c = 0; c < CT; ++c) {
            const u32x4* pb = (const u32x4*)(bRow0 + (size_t)c * 16 * K + k0);
            b[c].q[0] = pb[0];   // K = hi*16 + 0..7
            b[c].q[1] = pb[1];   // K = hi*16 + 8..15
        }
        #pragma unroll
        for (int c = 0; c < CT; ++c) {
            acc[c] = __builtin_amdgcn_wmma_f32_16x16x32_bf16(
                false, a.v, false, b[c].v, (short)0, acc[c], false, false);
        }
    }

    #pragma unroll
    for (int c = 0; c < CT; ++c) {
        bf16_t* out = H + ((size_t)(tM * 16 + hi * 8)) * ldH + (tg * CT + c) * 16 + r;
        #pragma unroll
        for (int j = 0; j < 8; ++j) out[(size_t)j * ldH] = (bf16_t)acc[c][j];
    }
}

// ---------- aggregation ----------
// Seed AGG with the self-loop term: AGG[i,f] = dinv[i]^2 * H[i,f]
__global__ void agg_init_kernel(const float* __restrict__ dinv, const bf16_t* __restrict__ H,
                                float* __restrict__ AGG, int total, int ld) {
    int i = blockIdx.x * blockDim.x + threadIdx.x;
    if (i >= total) return;
    int row = i / ld;
    float dv = dinv[row];
    AGG[i] = dv * dv * (float)H[i];
}

// Edge scatter-add: each thread handles one edge x 4 consecutive features.
__global__ void edge_agg_kernel(const long long* __restrict__ src,
                                const long long* __restrict__ dst,
                                const float* __restrict__ dinv,
                                const bf16_t* __restrict__ H,
                                float* __restrict__ AGG,
                                int e, int chunks, int ld) {
    int idx = blockIdx.x * blockDim.x + threadIdx.x;
    int ei = idx / chunks;
    if (ei >= e) return;
    int c = idx - ei * chunks;
    int s = (int)src[ei];
    int d = (int)dst[ei];
    float nrm = dinv[s] * dinv[d];
    u32x2 raw = *(const u32x2*)(H + (size_t)s * ld + c * 4);   // 4 bf16, 8 bytes
    float* ap = AGG + (size_t)d * ld + c * 4;
    float f0 = __uint_as_float(raw.x << 16)          * nrm;
    float f1 = __uint_as_float(raw.x & 0xffff0000u)  * nrm;
    float f2 = __uint_as_float(raw.y << 16)          * nrm;
    float f3 = __uint_as_float(raw.y & 0xffff0000u)  * nrm;
    atomicAdd(ap + 0, f0);
    atomicAdd(ap + 1, f1);
    atomicAdd(ap + 2, f2);
    atomicAdd(ap + 3, f3);
}

// Bias + ReLU fused with fp32 -> bf16 downcast for next layer input (F == 256).
__global__ void bias_relu_kernel(const float* __restrict__ AGG, const float* __restrict__ b,
                                 bf16_t* __restrict__ out, int total) {
    int i = blockIdx.x * blockDim.x + threadIdx.x;
    if (i >= total) return;
    float v = AGG[i] + b[i & 255];
    out[i] = (bf16_t)fmaxf(v, 0.0f);
}

// Final layer: bias + log_softmax over 40 classes per row (AGG row stride 48).
__global__ void final_logsoftmax_kernel(const float* __restrict__ AGG,
                                        const float* __restrict__ b,
                                        float* __restrict__ out, int n) {
    int r = blockIdx.x * blockDim.x + threadIdx.x;
    if (r >= n) return;
    const float* a = AGG + (size_t)r * 48;
    float v[40];
    float m = -1e30f;
    #pragma unroll
    for (int c = 0; c < 40; ++c) { v[c] = a[c] + b[c]; m = fmaxf(m, v[c]); }
    float s = 0.0f;
    #pragma unroll
    for (int c = 0; c < 40; ++c) s += __expf(v[c] - m);
    float ls = __logf(s) + m;
    float* o = out + (size_t)r * 40;
    #pragma unroll
    for (int c = 0; c < 40; ++c) o[c] = v[c] - ls;
}

// ---------- driver ----------
extern "C" void kernel_launch(void* const* d_in, const int* in_sizes, int n_in,
                              void* d_out, int out_size, void* d_ws, size_t ws_size,
                              hipStream_t stream) {
    const float*     x    = (const float*)d_in[0];
    const long long* edge = (const long long*)d_in[1];   // int64 [2, E] flat
    const float* W1 = (const float*)d_in[2]; const float* b1 = (const float*)d_in[3];
    const float* W2 = (const float*)d_in[4]; const float* b2 = (const float*)d_in[5];
    const float* W3 = (const float*)d_in[6]; const float* b3 = (const float*)d_in[7];
    const float* W4 = (const float*)d_in[8]; const float* b4 = (const float*)d_in[9];

    const int N = 100000, FIN = 128, HID = 256, NCLS = 40, CPAD = 48;
    const int E = in_sizes[1] / 2;
    const long long* src = edge;
    const long long* dst = edge + E;

    // Workspace carve-up (~206 MB total)
    char* wp = (char*)d_ws;
    auto carve = [&](size_t bytes) -> void* {
        void* p = (void*)wp;
        wp += (bytes + 255) & ~((size_t)255);
        return p;
    };
    float*  deg  = (float*)carve((size_t)N * 4);
    float*  dinv = (float*)carve((size_t)N * 4);
    bf16_t* Wt   = (bf16_t*)carve((size_t)HID * HID * 2);
    bf16_t* bufX = (bf16_t*)carve((size_t)N * HID * 2);
    bf16_t* bufH = (bf16_t*)carve((size_t)N * HID * 2);
    float*  AGG  = (float*)carve((size_t)N * HID * 4);
    (void)ws_size; (void)n_in; (void)out_size;

    const int T = 256;
    auto blocks = [&](long long t) { return (unsigned)((t + T - 1) / T); };

    // D^{-1/2}
    init_deg_kernel<<<blocks(N), T, 0, stream>>>(deg, N);
    deg_kernel<<<blocks(E), T, 0, stream>>>(dst, deg, E);
    dinv_kernel<<<blocks(N), T, 0, stream>>>(deg, dinv, N);

    // x -> bf16
    convert_x_kernel<<<blocks((long long)N * FIN), T, 0, stream>>>(x, bufX, N * FIN);

    auto layer = [&](const float* W, const float* b, int K, int Fout, int FoutPad, bool relu) {
        convert_w_kernel<<<blocks((long long)FoutPad * K), T, 0, stream>>>(W, Wt, K, Fout, FoutPad);
        int nRowTiles = N / 16;  // 6250 (exact)
        if (FoutPad == 256) {
            // 16 col tiles -> 4 groups of 4
            int nColGroups = 4;
            long long waves = (long long)nRowTiles * nColGroups;
            gemm_wmma_kernel<4><<<blocks(waves * 32), T, 0, stream>>>(
                bufX, Wt, bufH, nRowTiles, nColGroups, K, FoutPad);
        } else {
            // 48 padded cols -> 3 col tiles, one group
            int nColGroups = 1;
            long long waves = (long long)nRowTiles * nColGroups;
            gemm_wmma_kernel<3><<<blocks(waves * 32), T, 0, stream>>>(
                bufX, Wt, bufH, nRowTiles, nColGroups, K, FoutPad);
        }
        int total = N * FoutPad;
        agg_init_kernel<<<blocks(total), T, 0, stream>>>(dinv, bufH, AGG, total, FoutPad);
        int chunks = FoutPad / 4;
        edge_agg_kernel<<<blocks((long long)E * chunks), T, 0, stream>>>(
            src, dst, dinv, bufH, AGG, E, chunks, FoutPad);
        if (relu)
            bias_relu_kernel<<<blocks(total), T, 0, stream>>>(AGG, b, bufX, total);
    };

    layer(W1, b1, FIN, HID, HID, true);
    layer(W2, b2, HID, HID, HID, true);
    layer(W2, b2, HID, HID, HID, true);   // conv2 reused, as in reference
    layer(W3, b3, HID, HID, HID, true);
    layer(W4, b4, HID, NCLS, CPAD, false);

    final_logsoftmax_kernel<<<blocks(N), T, 0, stream>>>(AGG, b4, (float*)d_out, N);
}
